// SMC_Transf_Cell_60851096650039
// MI455X (gfx1250) — matrix operand order repaired and checked by reading, not compiled, not run on hardware
//
#include <hip/hip_runtime.h>
#include <hip/hip_bf16.h>
#include <math.h>

// ---------------- problem constants ----------------
constexpr int B_   = 8;
constexpr int P_   = 16;
constexpr int S_   = 512;
constexpr int D_   = 512;
constexpr int H_   = 8;
constexpr int DFF_ = 2048;
constexpr int VOC_ = 10000;
constexpr int DEP_ = 64;          // D/H
constexpr int BP_  = B_ * P_;     // 128 rows
constexpr float SIGMA_  = 0.1f;
constexpr float LN_EPS_ = 1e-6f;

// ---------------- output layout (floats, concatenated in return order) ----
constexpr size_t OFF_R4    = 0;                                   // (B,P,1,D)
constexpr size_t OFF_Z     = OFF_R4    + (size_t)BP_ * D_;        // (B,P,1,D)
constexpr size_t OFF_AVGSM = OFF_Z     + (size_t)BP_ * D_;        // (B,1,VOC)
constexpr size_t OFF_GOOD  = OFF_AVGSM + (size_t)B_ * VOC_;       // (B,1,VOC)
constexpr size_t OFF_MAXP  = OFF_GOOD  + (size_t)B_ * VOC_;       // (B,1,VOC)
constexpr size_t OFF_EPSZ  = OFF_MAXP  + (size_t)B_ * VOC_;       // (B,P,1,D)
constexpr size_t OFF_ATTNW = OFF_EPSZ  + (size_t)BP_ * D_;        // (B,P,1,S)
constexpr size_t OFF_K     = OFF_ATTNW + (size_t)BP_ * S_;        // (B,P,S,D)
constexpr size_t OFF_VS    = OFF_K     + (size_t)BP_ * S_ * D_;   // (B,P,S,D)
constexpr size_t OFF_W     = OFF_VS    + (size_t)BP_ * S_ * D_;   // (B,P,1)
constexpr size_t OFF_I     = OFF_W     + (size_t)BP_;             // (B,P,S)

// ---------------- workspace layout (floats) ----------------
constexpr size_t WS_Q     = 0;
constexpr size_t WS_K     = WS_Q     + (size_t)BP_ * D_;
constexpr size_t WS_V     = WS_K     + (size_t)BP_ * D_;
constexpr size_t WS_ZH    = WS_V     + (size_t)BP_ * D_;
constexpr size_t WS_ZP    = WS_ZH    + (size_t)BP_ * D_;
constexpr size_t WS_OUT1  = WS_ZP    + (size_t)BP_ * D_;
constexpr size_t WS_H1    = WS_OUT1  + (size_t)BP_ * D_;
constexpr size_t WS_FFN   = WS_H1    + (size_t)BP_ * DFF_;
constexpr size_t WS_RN    = WS_FFN   + (size_t)BP_ * D_;
constexpr size_t WS_ATTNP = WS_RN    + (size_t)BP_ * D_;          // (BP,H,S)
constexpr size_t WS_PRED  = WS_ATTNP + (size_t)BP_ * H_ * S_;
constexpr size_t WS_RMAX  = WS_PRED  + (size_t)BP_ * VOC_;
constexpr size_t WS_RSUM  = WS_RMAX  + BP_;
constexpr size_t WS_WSQ   = WS_RSUM  + BP_;
constexpr size_t WS_IT    = WS_WSQ   + BP_;   // int storage
constexpr size_t WS_JT    = WS_IT    + BP_;   // int storage
constexpr size_t WS_AMAX  = WS_JT    + BP_;   // int storage
constexpr size_t WS_END   = WS_AMAX  + 8;     // float count used

// ---- transposed f16 weight copies (offsets in _Float16 units, after WS_END)
constexpr size_t HT_Q   = 0;                                  // (D,D) col-major
constexpr size_t HT_K   = HT_Q   + (size_t)D_ * D_;
constexpr size_t HT_V   = HT_K   + (size_t)D_ * D_;
constexpr size_t HT_O   = HT_V   + (size_t)D_ * D_;
constexpr size_t HT_1   = HT_O   + (size_t)D_ * D_;           // (D,DFF) -> (DFF,D)
constexpr size_t HT_2   = HT_1   + (size_t)D_ * DFF_;         // (DFF,D) -> (D,DFF)
constexpr size_t HT_OUT = HT_2   + (size_t)DFF_ * D_;         // (D,VOC) -> (VOC,D)

// ---------------- vector types for WMMA ----------------
typedef _Float16 v16h __attribute__((ext_vector_type(16)));
typedef float    v8f  __attribute__((ext_vector_type(8)));

// ---------------- wave32 reductions ----------------
__device__ inline float wave_sum(float v) {
  #pragma unroll
  for (int o = 16; o > 0; o >>= 1) v += __shfl_xor(v, o, 32);
  return v;
}
__device__ inline float wave_max(float v) {
  #pragma unroll
  for (int o = 16; o > 0; o >>= 1) v = fmaxf(v, __shfl_xor(v, o, 32));
  return v;
}

// ---------------- weight convert + transpose: W[K,N] f32 -> Wt[N,K] f16 ----
__global__ void k_cvt_tr(const float* __restrict__ W, int K, int N,
                         _Float16* __restrict__ Wt) {
  const size_t e = (size_t)blockIdx.x * 256 + threadIdx.x;
  if (e >= (size_t)K * N) return;
  const int k = (int)(e / N), n = (int)(e % N);   // coalesced read along n
  Wt[(size_t)n * K + k] = (_Float16)W[e];
}

// ---------------- one 16x16 tile, K-loop of v_wmma_f32_16x16x32_f16 --------
// A: MxK row-major f32 (converted to f16 on load).
// Bt: NxK col-major-of-original = transposed f16 weights.
// ISA 7.12.2 wave32 layouts:
//   A 16x32 f16 : lane L -> row M=L%16 ; elems 0..7 -> K=(L/16)*8 + 0..7,
//                 elems 8..15 -> K=16+(L/16)*8 + 0..7
//   B 32x16 f16 : lane L -> col N=L%16 ; this code's order for A and B: elems e -> K=(L/16)*16 + e
//                 => 16 contiguous halves of Bt at (n0+L%16)*KDIM + kk + (L/16)*16
//   C/D 16x16 f32: VGPR r -> M=8*(L/16)+r, N=L%16
template <int KDIM>
__device__ inline void gemm_tile(const float* __restrict__ A,
                                 const _Float16* __restrict__ Bt,
                                 int m0, int n0, v8f& acc, int lane) {
  const int half = lane >> 4;     // 0 or 1
  const int lm   = lane & 15;
  const float*    arow = A  + (size_t)(m0 + lm) * KDIM + half * 16;
  const _Float16* brow = Bt + (size_t)(n0 + lm) * KDIM + half * 16;
  #pragma unroll 4
  for (int kk = 0; kk < KDIM; kk += 32) {
    v16h a;
    #pragma unroll
    for (int e = 0; e < 8; ++e) {
      a[e]     = (_Float16)arow[kk + e];        // K = kk + half*16 + e
      a[e + 8] = (_Float16)arow[kk + e + 8];    // K = kk + half*16 + 8 + e
    }
    const v16h bfrag = *(const v16h*)(brow + kk);   // 2x global_load_b128
    acc = __builtin_amdgcn_wmma_f32_16x16x32_f16(
        false, a, false, bfrag, (short)0, acc, false, false);
  }
}

// Generic GEMM: out[M,N] = A[M,KDIM] @ W[KDIM,N] + bias (+ sigma*eps)
// (+ resid) (relu), with W given as transposed f16 Bt[N,KDIM].
// One wave per 16x16 tile; grid = (N/16, M/16). Branch-free epilogue.
template <int KDIM, bool HAS_EPS, bool HAS_RES, bool RELU>
__global__ void k_gemm(const float* __restrict__ A,
                       const _Float16* __restrict__ Bt, int N,
                       const float* __restrict__ bias,
                       const float* __restrict__ eps,
                       const float* __restrict__ resid,
                       float* __restrict__ out) {
  const int n0 = blockIdx.x * 16, m0 = blockIdx.y * 16;
  const int lane = threadIdx.x;
  v8f acc = {};
  gemm_tile<KDIM>(A, Bt, m0, n0, acc, lane);
  const int half = lane >> 4, lm = lane & 15;
  #pragma unroll
  for (int r = 0; r < 8; ++r) {
    const int m = m0 + half * 8 + r;
    const int n = n0 + lm;
    const size_t idx = (size_t)m * N + n;
    float val = acc[r] + bias[n];
    if constexpr (HAS_EPS) val += SIGMA_ * eps[idx];
    if constexpr (HAS_RES) val += resid[idx];
    if constexpr (RELU)    val = fmaxf(val, 0.0f);
    out[idx] = val;
  }
}

// ---------------- attention: one wave per (b,p,h) ----------------
__global__ void k_attn(const float* __restrict__ Kin,
                       const float* __restrict__ Vin,
                       const float* __restrict__ qws,
                       const float* __restrict__ knew,
                       const float* __restrict__ vnew,
                       const int* __restrict__ tptr,
                       float* __restrict__ zh,
                       float* __restrict__ attnp) {
  const int h = blockIdx.x, p = blockIdx.y, b = blockIdx.z;
  const int lane = threadIdx.x;
  const int t = tptr[0];
  const int bp = b * P_ + p;
  __shared__ float qs[DEP_];
  __shared__ float sc[S_];
  const float* qrow = qws + (size_t)bp * D_ + h * DEP_;
  qs[lane]      = qrow[lane];
  qs[lane + 32] = qrow[lane + 32];
  __syncthreads();
  // scores (masked softmax over s<=t; s>t contributes exactly 0)
  float mx = -1e30f;
  for (int s = lane; s <= t; s += 32) {
    const float* krow = (s == t) ? (knew + (size_t)bp * D_ + h * DEP_)
                                 : (Kin + ((size_t)bp * S_ + s) * D_ + h * DEP_);
    float dot = 0.f;
    #pragma unroll
    for (int d = 0; d < DEP_; ++d) dot += qs[d] * krow[d];
    dot *= 0.125f;                   // 1/sqrt(DEPTH)
    sc[s] = dot;
    mx = fmaxf(mx, dot);
  }
  mx = wave_max(mx);
  float sum = 0.f;
  for (int s = lane; s <= t; s += 32) {
    const float e = expf(sc[s] - mx);
    sc[s] = e;
    sum += e;
  }
  sum = wave_sum(sum);
  const float inv = 1.0f / sum;
  for (int s = lane; s <= t; s += 32)
    attnp[((size_t)bp * H_ + h) * S_ + s] = sc[s] * inv;
  __syncthreads();
  // weighted V sum; lanes cover depth (coalesced across lanes)
  #pragma unroll
  for (int dd = 0; dd < 2; ++dd) {
    const int d = dd * 32 + lane;
    float accv = 0.f;
    for (int s = 0; s <= t; ++s) {
      const float* vrow = (s == t) ? (vnew + (size_t)bp * D_ + h * DEP_)
                                   : (Vin + ((size_t)bp * S_ + s) * D_ + h * DEP_);
      accv += sc[s] * vrow[d];
    }
    zh[(size_t)bp * D_ + h * DEP_ + d] = accv * inv;
  }
}

// ---------------- layernorm over last dim (D); one wave per row -----------
__global__ void k_addln(const float* __restrict__ a,
                        const float* __restrict__ addend,  // may be null
                        const float* __restrict__ g,
                        const float* __restrict__ beta,
                        float* __restrict__ out) {
  const int row = blockIdx.x, lane = threadIdx.x;
  const float* xa = a + (size_t)row * D_;
  const float* xb = addend ? addend + (size_t)row * D_ : nullptr;
  float s = 0.f;
  for (int i = lane; i < D_; i += 32) s += xa[i] + (xb ? xb[i] : 0.f);
  const float mu = wave_sum(s) * (1.0f / D_);
  float vs = 0.f;
  for (int i = lane; i < D_; i += 32) {
    const float d = xa[i] + (xb ? xb[i] : 0.f) - mu;
    vs += d * d;
  }
  const float var = wave_sum(vs) * (1.0f / D_);
  const float invs = rsqrtf(var + LN_EPS_);
  for (int i = lane; i < D_; i += 32) {
    const float xv = xa[i] + (xb ? xb[i] : 0.f);
    out[(size_t)row * D_ + i] = (xv - mu) * invs * g[i] + beta[i];
  }
}

// ---------------- per-row softmax stats over VOC ----------------
__global__ void k_predstats(const float* __restrict__ pred,
                            const int* __restrict__ x,
                            float* __restrict__ rmax,
                            float* __restrict__ rsum,
                            float* __restrict__ wsq) {
  const int row = blockIdx.x, tid = threadIdx.x;
  __shared__ float red[256];
  const float* pr = pred + (size_t)row * VOC_;
  float m = -1e30f;
  for (int i = tid; i < VOC_; i += 256) m = fmaxf(m, pr[i]);
  red[tid] = m; __syncthreads();
  for (int s = 128; s > 0; s >>= 1) {
    if (tid < s) red[tid] = fmaxf(red[tid], red[tid + s]);
    __syncthreads();
  }
  const float mx = red[0]; __syncthreads();
  float sm = 0.f;
  for (int i = tid; i < VOC_; i += 256) sm += expf(pr[i] - mx);
  red[tid] = sm; __syncthreads();
  for (int s = 128; s > 0; s >>= 1) {
    if (tid < s) red[tid] += red[tid + s];
    __syncthreads();
  }
  if (tid == 0) {
    const float sum = red[0];
    rmax[row] = mx;
    rsum[row] = sum;
    const int tok = x[row / P_];
    wsq[row] = expf(pr[tok] - mx) / sum;
  }
}

// ---------------- gumbel argmax resampling indices; one wave per b --------
__global__ void k_sample(const float* __restrict__ wsq,
                         const float* __restrict__ gumbel,
                         int* __restrict__ i_t, int* __restrict__ j_t,
                         int* __restrict__ amax) {
  const int b = blockIdx.x, lane = threadIdx.x;
  __shared__ int it_s[P_];
  if (lane < P_) {
    float best = -1e30f; int bi = 0;
    for (int pp = 0; pp < P_; ++pp) {
      const float val = logf(wsq[b * P_ + pp] + 1e-10f) +
                        gumbel[((size_t)(b * P_ + lane)) * P_ + pp];
      if (val > best) { best = val; bi = pp; }   // first max, like jnp.argmax
    }
    it_s[lane] = bi;
    i_t[b * P_ + lane] = bi;
  }
  if (lane == 0) {
    float best = -1e30f; int bi = 0;
    for (int pp = 0; pp < P_; ++pp) {
      const float v = wsq[b * P_ + pp];
      if (v > best) { best = v; bi = pp; }
    }
    amax[b] = bi;
  }
  __syncthreads();
  if (lane < P_) j_t[b * P_ + lane] = it_s[it_s[lane]];
}

// ---------------- per-batch vocab means / max_prediction ----------------
__global__ void k_predavg(const float* __restrict__ pred,
                          const float* __restrict__ rmax,
                          const float* __restrict__ rsum,
                          const int* __restrict__ amax,
                          float* __restrict__ out_avgsm,
                          float* __restrict__ out_good,
                          float* __restrict__ out_maxp) {
  const int b = blockIdx.y;
  const int v = blockIdx.x * 256 + threadIdx.x;
  if (v >= VOC_) return;
  float ga = 0.f, as = 0.f;
  for (int pp = 0; pp < P_; ++pp) {
    const int row = b * P_ + pp;
    const float pv = pred[(size_t)row * VOC_ + v];
    ga += pv;
    as += expf(pv - rmax[row]) / rsum[row];
  }
  out_good[(size_t)b * VOC_ + v]  = ga * (1.0f / P_);
  out_avgsm[(size_t)b * VOC_ + v] = as * (1.0f / P_);
  out_maxp[(size_t)b * VOC_ + v]  =
      pred[(size_t)(b * P_ + amax[b]) * VOC_ + v];
}

// ---------------- K / Vs resample: float4 row copies (the bandwidth term) -
__global__ void k_resample(const float* __restrict__ Kin,
                           const float* __restrict__ knew,
                           const int* __restrict__ i_t,
                           const int* __restrict__ j_t,
                           const int* __restrict__ tptr,
                           float* __restrict__ Kout,
                           float* __restrict__ Vout) {
  const int s = blockIdx.x, p = blockIdx.y, b = blockIdx.z;
  const int tid = threadIdx.x;            // 128 threads, 4 floats each
  const int t = tptr[0];
  const int bp = b * P_ + p;
  const int pi = i_t[bp], pj = j_t[bp];
  const float4* srcK; const float4* srcV;
  if (s < t) {
    srcK = (const float4*)(Kin + ((size_t)(b * P_ + pi) * S_ + s) * D_);
    srcV = (const float4*)(Kin + ((size_t)(b * P_ + pj) * S_ + s) * D_);
  } else if (s == t) {
    srcK = (const float4*)(knew + (size_t)(b * P_ + pi) * D_);
    srcV = (const float4*)(knew + (size_t)(b * P_ + pj) * D_);
  } else {
    srcK = (const float4*)(Kin + ((size_t)bp * S_ + s) * D_);
    srcV = srcK;
  }
  // hint the streaming gather (global_prefetch_b8 on gfx1250)
  __builtin_prefetch((const void*)(srcK + tid), 0, 0);
  __builtin_prefetch((const void*)(srcV + tid), 0, 0);
  float4* dK = (float4*)(Kout + ((size_t)bp * S_ + s) * D_);
  float4* dV = (float4*)(Vout + ((size_t)bp * S_ + s) * D_);
  dK[tid] = srcK[tid];
  dV[tid] = srcV[tid];
}

// ---------------- remaining small outputs ----------------
__global__ void k_misc(const float* __restrict__ r,
                       const float* __restrict__ eps_z,
                       const float* __restrict__ zws,
                       const float* __restrict__ attnp,
                       const int* __restrict__ Iin,
                       const int* __restrict__ i_t,
                       const float* __restrict__ wsq,
                       const int* __restrict__ tptr,
                       float* __restrict__ out) {
  const int e = blockIdx.x * 256 + threadIdx.x;   // 0 .. 65535
  const int t = tptr[0];
  out[OFF_R4 + e]   = r[e];
  out[OFF_EPSZ + e] = eps_z[e];
  // attn_weights = mean over heads (zero where s > t)
  {
    const int bp = e / S_, s = e % S_;
    float aw = 0.f;
    if (s <= t) {
      #pragma unroll
      for (int h = 0; h < H_; ++h)
        aw += attnp[((size_t)bp * H_ + h) * S_ + s];
      aw *= (1.0f / H_);
    }
    out[OFF_ATTNW + e] = aw;
  }
  // I_new
  {
    const int b = e / (P_ * S_), rem = e % (P_ * S_);
    const int p = rem / S_, s = rem % S_;
    int val;
    if (s < t)       val = Iin[(size_t)(b * P_ + i_t[b * P_ + p]) * S_ + s];
    else if (s == t) val = i_t[b * P_ + p];
    else             val = Iin[e];
    out[OFF_I + e] = (float)val;
  }
  // z gathered by I_new[:,:,t] == i_t
  {
    const int b = e / (P_ * D_);
    const int p = (e / D_) % P_;
    const int d = e % D_;
    out[OFF_Z + e] = zws[(size_t)(b * P_ + i_t[b * P_ + p]) * D_ + d];
  }
  if (e < BP_) out[OFF_W + e] = wsq[e];
}

// ---------------- host side ----------------
extern "C" void kernel_launch(void* const* d_in, const int* in_sizes, int n_in,
                              void* d_out, int out_size, void* d_ws, size_t ws_size,
                              hipStream_t stream) {
  (void)in_sizes; (void)n_in; (void)out_size; (void)ws_size;
  const float* r     = (const float*)d_in[0];
  const int*   x     = (const int*)  d_in[1];
  const float* Kin   = (const float*)d_in[2];
  const float* Vin   = (const float*)d_in[3];
  const int*   Iin   = (const int*)  d_in[5];
  const int*   tptr  = (const int*)  d_in[6];
  const float* eps_q = (const float*)d_in[7];
  const float* eps_k = (const float*)d_in[8];
  const float* eps_v = (const float*)d_in[9];
  const float* eps_z = (const float*)d_in[10];
  const float* gumb  = (const float*)d_in[11];
  const float* Wq    = (const float*)d_in[12];
  const float* bq    = (const float*)d_in[13];
  const float* Wk    = (const float*)d_in[14];
  const float* bk    = (const float*)d_in[15];
  const float* Wv    = (const float*)d_in[16];
  const float* bv    = (const float*)d_in[17];
  const float* Wo    = (const float*)d_in[18];
  const float* bo    = (const float*)d_in[19];
  const float* ln1g  = (const float*)d_in[20];
  const float* ln1b  = (const float*)d_in[21];
  const float* ln3g  = (const float*)d_in[22];
  const float* ln3b  = (const float*)d_in[23];
  const float* W1    = (const float*)d_in[24];
  const float* b1    = (const float*)d_in[25];
  const float* W2    = (const float*)d_in[26];
  const float* b2    = (const float*)d_in[27];
  const float* Wout  = (const float*)d_in[28];
  const float* bout  = (const float*)d_in[29];

  float* ws   = (float*)d_ws;
  float* wq   = ws + WS_Q;
  float* wk   = ws + WS_K;
  float* wv   = ws + WS_V;
  float* zh   = ws + WS_ZH;
  float* zp   = ws + WS_ZP;
  float* out1 = ws + WS_OUT1;
  float* h1   = ws + WS_H1;
  float* ffn  = ws + WS_FFN;
  float* rn   = ws + WS_RN;
  float* atp  = ws + WS_ATTNP;
  float* pred = ws + WS_PRED;
  float* rmax = ws + WS_RMAX;
  float* rsum = ws + WS_RSUM;
  float* wsq  = ws + WS_WSQ;
  int*   i_t  = (int*)(ws + WS_IT);
  int*   j_t  = (int*)(ws + WS_JT);
  int*   amax = (int*)(ws + WS_AMAX);
  _Float16* hws = (_Float16*)(ws + WS_END);
  _Float16* tq  = hws + HT_Q;
  _Float16* tk  = hws + HT_K;
  _Float16* tv  = hws + HT_V;
  _Float16* to_ = hws + HT_O;
  _Float16* t1  = hws + HT_1;
  _Float16* t2  = hws + HT_2;
  _Float16* tou = hws + HT_OUT;
  float* out  = (float*)d_out;

  const dim3 wave(32);
  const dim3 gD(D_ / 16, BP_ / 16);
  auto nblk = [](size_t n) { return dim3((unsigned)((n + 255) / 256)); };

  // one-time (per launch) weight convert+transpose to f16 col-major
  k_cvt_tr<<<nblk((size_t)D_ * D_),   256, 0, stream>>>(Wq,   D_,   D_,   tq);
  k_cvt_tr<<<nblk((size_t)D_ * D_),   256, 0, stream>>>(Wk,   D_,   D_,   tk);
  k_cvt_tr<<<nblk((size_t)D_ * D_),   256, 0, stream>>>(Wv,   D_,   D_,   tv);
  k_cvt_tr<<<nblk((size_t)D_ * D_),   256, 0, stream>>>(Wo,   D_,   D_,   to_);
  k_cvt_tr<<<nblk((size_t)D_ * DFF_), 256, 0, stream>>>(W1,   D_,   DFF_, t1);
  k_cvt_tr<<<nblk((size_t)DFF_ * D_), 256, 0, stream>>>(W2,   DFF_, D_,   t2);
  k_cvt_tr<<<nblk((size_t)D_ * VOC_), 256, 0, stream>>>(Wout, D_,   VOC_, tou);

  // QKV projections (WMMA, f32 A -> f16, f16 transposed weights): 128x512x512
  k_gemm<D_, true,  false, false><<<gD, wave, 0, stream>>>(r, tq, D_, bq, eps_q, nullptr, wq);
  k_gemm<D_, true,  false, false><<<gD, wave, 0, stream>>>(r, tk, D_, bk, eps_k, nullptr, wk);
  k_gemm<D_, true,  false, false><<<gD, wave, 0, stream>>>(r, tv, D_, bv, eps_v, nullptr, wv);

  // attention per (b,p,h)
  k_attn<<<dim3(H_, P_, B_), wave, 0, stream>>>(Kin, Vin, wq, wk, wv, tptr, zh, atp);

  // z = zh @ Wo + bo + sigma*eps_z
  k_gemm<D_, true,  false, false><<<gD, wave, 0, stream>>>(zh, to_, D_, bo, eps_z, nullptr, zp);

  // out1 = LN(z + r)
  k_addln<<<dim3(BP_), wave, 0, stream>>>(zp, r, ln1g, ln1b, out1);

  // FFN: relu(out1@W1+b1) @ W2 + b2 + out1, then LN
  k_gemm<D_,   false, false, true ><<<dim3(DFF_ / 16, BP_ / 16), wave, 0, stream>>>(
      out1, t1, DFF_, b1, nullptr, nullptr, h1);
  k_gemm<DFF_, false, true,  false><<<gD, wave, 0, stream>>>(
      h1, t2, D_, b2, nullptr, out1, ffn);
  k_addln<<<dim3(BP_), wave, 0, stream>>>(ffn, nullptr, ln3g, ln3b, rn);

  // predictions = rn @ Wout + bout  (128 x 512 x 10000)
  k_gemm<D_, false, false, false><<<dim3(VOC_ / 16, BP_ / 16), wave, 0, stream>>>(
      rn, tou, VOC_, bout, nullptr, nullptr, pred);

  // softmax stats + w_sq
  k_predstats<<<dim3(BP_), dim3(256), 0, stream>>>(pred, x, rmax, rsum, wsq);

  // gumbel resampling indices
  k_sample<<<dim3(B_), wave, 0, stream>>>(wsq, gumb, i_t, j_t, amax);

  // per-batch vocab outputs (directly into d_out)
  k_predavg<<<dim3((VOC_ + 255) / 256, B_), dim3(256), 0, stream>>>(
      pred, rmax, rsum, amax, out + OFF_AVGSM, out + OFF_GOOD, out + OFF_MAXP);

  // K / Vs resample (bandwidth-dominant: 2 x 134 MB stores)
  k_resample<<<dim3(S_, P_, B_), dim3(128), 0, stream>>>(
      Kin, wk, i_t, j_t, tptr, out + OFF_K, out + OFF_VS);

  // remaining outputs
  k_misc<<<dim3((BP_ * S_) / 256), dim3(256), 0, stream>>>(
      r, eps_z, zp, atp, Iin, i_t, wsq, tptr, out);
}
